// HUBNETBlockV2_28544352649674
// MI455X (gfx1250) — matrix operand order repaired and checked by reading, not compiled, not run on hardware
//
#include <hip/hip_runtime.h>

// ---------------------------------------------------------------------------
// HUBNET slot-memory block for MI455X (gfx1250), wave32 + WMMA bf16.
// All matmuls run through v_wmma_f32_16x16x32_bf16. Each wave computes a
// 32x32 register block (2x2 tiles -> 4 independent WMMA accumulator chains,
// ~16 FLOP/byte of fragment traffic). Weights are converted once to bf16 and
// transposed to [N,K] so both A and B fragments are contiguous 16B loads
// (same contiguous k = 16*half + s order in A and B). Speculative global_prefetch_b8 issued unconditionally
// (OOB speculative prefetch is dropped by hardware; no branch guards needed).
// ---------------------------------------------------------------------------

typedef __attribute__((ext_vector_type(16))) __bf16 v16bf;
typedef __attribute__((ext_vector_type(8)))  __bf16 v8bf;
typedef __attribute__((ext_vector_type(8)))  float  v8f;

static constexpr int Bc = 16, Tc = 2048, Dc = 512, Sc = 64, Cc = 8;
static constexpr int STEPS = 3;
static constexpr float CLIPV = 10.0f;

__device__ __forceinline__ __bf16 f2bf(float f) {
  unsigned u = __builtin_bit_cast(unsigned, f);
  unsigned r = u + 0x7FFFu + ((u >> 16) & 1u);   // round-to-nearest-even
  unsigned short h = (unsigned short)(r >> 16);
  return __builtin_bit_cast(__bf16, h);
}
__device__ __forceinline__ float bf2f(__bf16 b) {
  unsigned short h = __builtin_bit_cast(unsigned short, b);
  unsigned u = ((unsigned)h) << 16;
  return __builtin_bit_cast(float, u);
}

__device__ __forceinline__ float block_reduce_sum_256(float v) {
  __shared__ float buf[256];
  int tid = threadIdx.x;
  buf[tid] = v;
  __syncthreads();
  for (int s = 128; s > 0; s >>= 1) {
    if (tid < s) buf[tid] += buf[tid + s];
    __syncthreads();
  }
  float r = buf[0];
  __syncthreads();
  return r;
}

// ---------------------------------------------------------------------------
// WMMA GEMM: one 32x32 output block per wave (2x2 of 16x16 WMMA tiles),
// 8 waves (256 thr) per block.
// MODE 0: C[M,N] = A[M,K] * BT[N,K]^T          (A row-major, B pre-transposed)
// MODE 1: C[M,N] = A[K,M]^T * B[K,N]           (write einsum: binding^T @ v)
// MODE 2: C[M,N] = A[M,K]   * B[K,N]           (context: rw @ rv)
// epi: 0 none, 1 clip(+-10) after alpha, 2 sigmoid
// Requires M % 32 == 0, N % 32 == 0, K % 32 == 0 (true for all call sites).
// ---------------------------------------------------------------------------
template <int MODE>
__global__ void gemm_wmma_bf16(const __bf16* __restrict__ A, long aBatch,
                               const __bf16* __restrict__ Bm, long bBatch,
                               const float* __restrict__ bias,
                               float* __restrict__ outF,
                               __bf16* __restrict__ outBf, long cBatch,
                               int M, int N, int K, float alpha, int epi) {
  const int lane = threadIdx.x & 31;
  const int wave = threadIdx.x >> 5;
  const int n0 = (blockIdx.x * 8 + wave) * 32;
  const int m0 = blockIdx.y * 32;
  if (n0 >= N || m0 >= M) return;
  const int half = lane >> 4;   // lanes 16..31 handle the other K sub-run
  const int l = lane & 15;
  const __bf16* Ab = A + (long)blockIdx.z * aBatch;
  const __bf16* Bb = Bm + (long)blockIdx.z * bBatch;
  const int arow0 = m0 + l, arow1 = m0 + 16 + l;
  const int bcol0 = n0 + l, bcol1 = n0 + 16 + l;

  v8f acc00 = {}, acc01 = {}, acc10 = {}, acc11 = {};
  for (int k0 = 0; k0 < K; k0 += 32) {
    v16bf a0v, a1v, b0v, b1v;
    // ---- A fragments: 16x32 A, contiguous k = 16*half + s per lane ----
    if constexpr (MODE == 0 || MODE == 2) {
      const __bf16* ap0 = Ab + (long)arow0 * K + k0 + (half ? 16 : 0);
      const __bf16* ap1 = Ab + (long)arow1 * K + k0 + (half ? 16 : 0);
      v8bf x0 = *reinterpret_cast<const v8bf*>(ap0);
      v8bf x1 = *reinterpret_cast<const v8bf*>(ap0 + 8);
      v8bf y0 = *reinterpret_cast<const v8bf*>(ap1);
      v8bf y1 = *reinterpret_cast<const v8bf*>(ap1 + 8);
      __builtin_prefetch(ap0 + 128, 0, 1);   // speculative; OOB is dropped
      __builtin_prefetch(ap1 + 128, 0, 1);
#pragma unroll
      for (int i = 0; i < 8; ++i) {
        a0v[i] = x0[i]; a0v[i + 8] = x1[i];
        a1v[i] = y0[i]; a1v[i + 8] = y1[i];
      }
    } else {  // MODE 1: A stored [K,M]; gather strided
      const int kb = k0 + (half ? 16 : 0);
#pragma unroll
      for (int i = 0; i < 8; ++i) {
        a0v[i]     = Ab[(long)(kb + i) * M + arow0];
        a0v[i + 8] = Ab[(long)(kb + 8 + i) * M + arow0];
        a1v[i]     = Ab[(long)(kb + i) * M + arow1];
        a1v[i + 8] = Ab[(long)(kb + 8 + i) * M + arow1];
      }
    }
    // ---- B fragments: 16-bit B 32x16 layout ----
    if constexpr (MODE == 0) {  // BT[N,K]: contiguous 16 K-values per lane
      const __bf16* bp0 = Bb + (long)bcol0 * K + k0 + (half ? 16 : 0);
      const __bf16* bp1 = Bb + (long)bcol1 * K + k0 + (half ? 16 : 0);
      v8bf x0 = *reinterpret_cast<const v8bf*>(bp0);
      v8bf x1 = *reinterpret_cast<const v8bf*>(bp0 + 8);
      v8bf y0 = *reinterpret_cast<const v8bf*>(bp1);
      v8bf y1 = *reinterpret_cast<const v8bf*>(bp1 + 8);
      __builtin_prefetch(bp0 + 128, 0, 1);
      __builtin_prefetch(bp1 + 128, 0, 1);
#pragma unroll
      for (int i = 0; i < 8; ++i) {
        b0v[i] = x0[i]; b0v[i + 8] = x1[i];
        b1v[i] = y0[i]; b1v[i + 8] = y1[i];
      }
    } else {  // B row-major [K,N]: strided gather
      const int kb = k0 + (half ? 16 : 0);
#pragma unroll
      for (int i = 0; i < 16; ++i) {
        b0v[i] = Bb[(long)(kb + i) * N + bcol0];
        b1v[i] = Bb[(long)(kb + i) * N + bcol1];
      }
    }
    // 4 independent accumulator chains -> pipelined XDL issue
    acc00 = __builtin_amdgcn_wmma_f32_16x16x32_bf16(false, a0v, false, b0v,
                                                    (short)0, acc00, false, false);
    acc01 = __builtin_amdgcn_wmma_f32_16x16x32_bf16(false, a0v, false, b1v,
                                                    (short)0, acc01, false, false);
    acc10 = __builtin_amdgcn_wmma_f32_16x16x32_bf16(false, a1v, false, b0v,
                                                    (short)0, acc10, false, false);
    acc11 = __builtin_amdgcn_wmma_f32_16x16x32_bf16(false, a1v, false, b1v,
                                                    (short)0, acc11, false, false);
  }
  // ---- epilogue: f32 C/D 16x16 layout: lane<16 rows 0..7, lane>=16 rows 8..15
  const float bb0 = bias ? bias[bcol0] : 0.0f;
  const float bb1 = bias ? bias[bcol1] : 0.0f;
  float* of = outF ? outF + (long)blockIdx.z * cBatch : nullptr;
  __bf16* ob = outBf ? outBf + (long)blockIdx.z * cBatch : nullptr;
  const v8f* accs[2][2] = {{&acc00, &acc01}, {&acc10, &acc11}};
#pragma unroll
  for (int ti = 0; ti < 2; ++ti) {
#pragma unroll
    for (int tj = 0; tj < 2; ++tj) {
      const v8f a = *accs[ti][tj];
      const int col = (tj ? bcol1 : bcol0);
      const float bb = (tj ? bb1 : bb0);
#pragma unroll
      for (int v = 0; v < 8; ++v) {
        const int row = m0 + 16 * ti + (half ? 8 : 0) + v;
        float r = a[v] * alpha + bb;
        if (epi == 1) r = fminf(fmaxf(r, -CLIPV), CLIPV);
        else if (epi == 2) r = 1.0f / (1.0f + __expf(-r));
        const long idx = (long)row * N + col;
        if (of) of[idx] = r;
        if (ob) ob[idx] = f2bf(r);
      }
    }
  }
}

// ---------------------------------------------------------------------------
// Weight convert + transpose: Wt[n*K + k] = bf16(W[k*N + n])
// ---------------------------------------------------------------------------
__global__ void w_transpose_bf16(const float* __restrict__ W,
                                 __bf16* __restrict__ Wt, int K, int N) {
  long i = (long)blockIdx.x * blockDim.x + threadIdx.x;
  if (i >= (long)K * N) return;
  int k = (int)(i / N), n = (int)(i % N);
  Wt[(long)n * K + k] = f2bf(W[i]);
}

// tokens = LN(token_emb[id] + pos_emb[t]) -> bf16. One 256-thr block per row.
__global__ void embed_ln_kernel(const int* __restrict__ ids,
                                const float* __restrict__ tokE,
                                const float* __restrict__ posE,
                                const float* __restrict__ g,
                                const float* __restrict__ bl,
                                __bf16* __restrict__ out) {
  const long row = blockIdx.x;            // b*T + t
  const int t = (int)(row % Tc);
  const int id = ids[row];
  const int d0 = threadIdx.x, d1 = threadIdx.x + 256;
  float x0 = tokE[(long)id * Dc + d0] + posE[(long)t * Dc + d0];
  float x1 = tokE[(long)id * Dc + d1] + posE[(long)t * Dc + d1];
  float mean = block_reduce_sum_256(x0 + x1) * (1.0f / Dc);
  float u0 = x0 - mean, u1 = x1 - mean;
  float var = block_reduce_sum_256(u0 * u0 + u1 * u1) * (1.0f / Dc);
  float rs = rsqrtf(var + 1e-5f);
  out[row * Dc + d0] = f2bf(u0 * rs * g[d0] + bl[d0]);
  out[row * Dc + d1] = f2bf(u1 * rs * g[d1] + bl[d1]);
}

// memory = broadcast(memory_init) (f32 + bf16)
__global__ void mem_init_kernel(const float* __restrict__ mi,
                                float* __restrict__ mem,
                                __bf16* __restrict__ memb, long total) {
  long i = (long)blockIdx.x * blockDim.x + threadIdx.x;
  if (i >= total) return;
  float v = mi[i % ((long)Sc * Dc)];
  mem[i] = v;
  memb[i] = f2bf(v);
}

// top-8 softmax over S=64 + masked norm. One wave per (b,t) row; 2 vals/lane.
__global__ void topk_softmax_kernel(const float* __restrict__ scores,
                                    const int* __restrict__ amask,
                                    __bf16* __restrict__ binding, long rows) {
  const int lane = threadIdx.x & 31;
  const int wave = threadIdx.x >> 5;
  const long row = (long)blockIdx.x * 8 + wave;
  if (row >= rows) return;
  const float* p = scores + row * Sc;
  float v0 = p[lane], v1 = p[lane + 32];
  float w0 = v0, w1 = v1;
  bool keep0 = false, keep1 = false;
  float rowmax = -1e30f;
  for (int it = 0; it < 8; ++it) {
    float bv; int bi;
    if (w0 >= w1) { bv = w0; bi = lane; } else { bv = w1; bi = lane + 32; }
    for (int off = 16; off; off >>= 1) {
      float ov = __shfl_xor(bv, off, 32);
      int oi = __shfl_xor(bi, off, 32);
      if (ov > bv || (ov == bv && oi < bi)) { bv = ov; bi = oi; }
    }
    if (it == 0) rowmax = bv;
    if (bi == lane)           { keep0 = true; w0 = -1e30f; }
    else if (bi == lane + 32) { keep1 = true; w1 = -1e30f; }
  }
  float e0 = keep0 ? __expf(v0 - rowmax) : 0.0f;
  float e1 = keep1 ? __expf(v1 - rowmax) : 0.0f;
  float s = e0 + e1;
  for (int off = 16; off; off >>= 1) s += __shfl_xor(s, off, 32);
  const float mk = (float)amask[row];
  float b0 = e0 / s * mk, b1 = e1 / s * mk;
  float den = fmaxf(mk, 1e-9f);  // sum_s(softmax*mask) == mk
  binding[row * Sc + lane]      = f2bf(b0 / den);
  binding[row * Sc + lane + 32] = f2bf(b1 / den);
}

// usage[b,s] = sum_t binding[b,t,s]; one 256-thr block per (b,s)
__global__ void slot_usage_kernel(const __bf16* __restrict__ binding,
                                  float* __restrict__ usage) {
  const int bs = blockIdx.x;
  const int b = bs / Sc, s = bs % Sc;
  float acc = 0.0f;
  for (int t = threadIdx.x; t < Tc; t += 256)
    acc += bf2f(binding[((long)b * Tc + t) * Sc + s]);
  float tot = block_reduce_sum_256(acc);
  if (threadIdx.x == 0) usage[bs] = tot;
}

// out_bf[i] = bf16(write[i] / max(usage[row], 1e-6))
__global__ void write_over_usage_kernel(const float* __restrict__ wr,
                                        const float* __restrict__ usage,
                                        __bf16* __restrict__ out, long total) {
  long i = (long)blockIdx.x * blockDim.x + threadIdx.x;
  if (i >= total) return;
  float u = fmaxf(usage[i / Dc], 1e-6f);
  out[i] = f2bf(wr[i] / u);
}

// memory = LN(memory + gate*write2) -> f32 + bf16. One block per (b,s) row.
__global__ void mem_update_ln_kernel(float* __restrict__ memory,
                                     __bf16* __restrict__ memb,
                                     const float* __restrict__ gate,
                                     const float* __restrict__ wr2,
                                     const float* __restrict__ g,
                                     const float* __restrict__ bl) {
  const long row = blockIdx.x;  // b*S + s
  const int d0 = threadIdx.x, d1 = threadIdx.x + 256;
  const long i0 = row * Dc + d0, i1 = row * Dc + d1;
  float x0 = memory[i0] + gate[i0] * wr2[i0];
  float x1 = memory[i1] + gate[i1] * wr2[i1];
  float mean = block_reduce_sum_256(x0 + x1) * (1.0f / Dc);
  float u0 = x0 - mean, u1 = x1 - mean;
  float var = block_reduce_sum_256(u0 * u0 + u1 * u1) * (1.0f / Dc);
  float rs = rsqrtf(var + 1e-5f);
  float r0 = u0 * rs * g[d0] + bl[d0];
  float r1 = u1 * rs * g[d1] + bl[d1];
  memory[i0] = r0; memory[i1] = r1;
  memb[i0] = f2bf(r0); memb[i1] = f2bf(r1);
}

// concat along features: out[row, 0:D]=a, [D:2D]=c
__global__ void pack2_bf_kernel(const __bf16* __restrict__ a,
                                const __bf16* __restrict__ c,
                                __bf16* __restrict__ out, long total) {
  long i = (long)blockIdx.x * blockDim.x + threadIdx.x;
  if (i >= total) return;
  long row = i / (2 * Dc); int col = (int)(i % (2 * Dc));
  out[i] = (col < Dc) ? a[row * Dc + col] : c[row * Dc + col - Dc];
}

// pooled[b,d] = sum_t fused*mask / max(sum_t mask, 1)
__global__ void pool_kernel(const float* __restrict__ fused,
                            const int* __restrict__ amask,
                            float* __restrict__ pooled) {
  const int chunk = blockIdx.x % (Dc / 256);
  const int b = blockIdx.x / (Dc / 256);
  const int d = chunk * 256 + threadIdx.x;
  float s = 0.0f, cm = 0.0f;
  for (int t = 0; t < Tc; ++t) {
    float mk = (float)amask[(long)b * Tc + t];
    s += fused[((long)b * Tc + t) * Dc + d] * mk;
    cm += mk;
  }
  pooled[(long)b * Dc + d] = s / fmaxf(cm, 1.0f);
}

// logits[b,c] = pooled[b,:] . cls_W[:,c] + cls_b[c]   (tiny)
__global__ void logits_kernel(const float* __restrict__ pooled,
                              const float* __restrict__ clsW,
                              const float* __restrict__ clsB,
                              float* __restrict__ out) {
  int i = blockIdx.x * blockDim.x + threadIdx.x;
  if (i >= Bc * Cc) return;
  int b = i / Cc, c = i % Cc;
  float s = clsB[c];
  for (int d = 0; d < Dc; ++d) s += pooled[(long)b * Dc + d] * clsW[(long)d * Cc + c];
  out[i] = s;
}

// ---------------------------------------------------------------------------
extern "C" void kernel_launch(void* const* d_in, const int* in_sizes, int n_in,
                              void* d_out, int out_size, void* d_ws, size_t ws_size,
                              hipStream_t stream) {
  // d_in order = setup_inputs() dict order (params dict flattened in order)
  const int*   input_ids  = (const int*)d_in[0];
  const int*   amask      = (const int*)d_in[1];
  const float* token_emb  = (const float*)d_in[2];
  const float* pos_emb    = (const float*)d_in[3];
  const float* memory_init= (const float*)d_in[4];
  const float* q_W  = (const float*)d_in[5];  const float* q_b  = (const float*)d_in[6];
  const float* k_W  = (const float*)d_in[7];  const float* k_b  = (const float*)d_in[8];
  const float* v_W  = (const float*)d_in[9];  const float* v_b  = (const float*)d_in[10];
  const float* wr_W = (const float*)d_in[11]; const float* wr_b = (const float*)d_in[12];
  const float* gate_W = (const float*)d_in[13]; const float* gate_b = (const float*)d_in[14];
  const float* rq_W = (const float*)d_in[15]; const float* rq_b = (const float*)d_in[16];
  const float* rk_W = (const float*)d_in[17]; const float* rk_b = (const float*)d_in[18];
  const float* rv_W = (const float*)d_in[19]; const float* rv_b = (const float*)d_in[20];
  const float* fuse_W = (const float*)d_in[21]; const float* fuse_b = (const float*)d_in[22];
  const float* tok_g = (const float*)d_in[23]; const float* tok_bb = (const float*)d_in[24];
  const float* slot_g = (const float*)d_in[25]; const float* slot_bb = (const float*)d_in[26];
  const float* cls_W = (const float*)d_in[27]; const float* cls_b = (const float*)d_in[28];
  float* logits = (float*)d_out;

  const long BT = (long)Bc * Tc, BS = (long)Bc * Sc;
  char* ws = (char*)d_ws;
  size_t off = 0;
  auto alloc = [&](size_t bytes) -> void* {
    void* p = ws + off;
    off = (off + bytes + 255) & ~(size_t)255;
    return p;
  };

  __bf16* tokens_bf  = (__bf16*)alloc(BT * Dc * 2);
  __bf16* qWt  = (__bf16*)alloc((size_t)Dc * Dc * 2);
  __bf16* kWt  = (__bf16*)alloc((size_t)Dc * Dc * 2);
  __bf16* vWt  = (__bf16*)alloc((size_t)Dc * Dc * 2);
  __bf16* wrWt = (__bf16*)alloc((size_t)Dc * Dc * 2);
  __bf16* rqWt = (__bf16*)alloc((size_t)Dc * Dc * 2);
  __bf16* rkWt = (__bf16*)alloc((size_t)Dc * Dc * 2);
  __bf16* rvWt = (__bf16*)alloc((size_t)Dc * Dc * 2);
  __bf16* gateWt = (__bf16*)alloc((size_t)2 * Dc * Dc * 2);
  __bf16* fuseWt = (__bf16*)alloc((size_t)2 * Dc * Dc * 2);
  __bf16* q_bf  = (__bf16*)alloc(BT * Dc * 2);
  __bf16* v_bf  = (__bf16*)alloc(BT * Dc * 2);
  __bf16* rq_bf = (__bf16*)alloc(BT * Dc * 2);
  float*  memory_f  = (float*)alloc(BS * Dc * 4);
  __bf16* memory_bf = (__bf16*)alloc(BS * Dc * 2);
  __bf16* k_bf      = (__bf16*)alloc(BS * Dc * 2);
  float*  scores_f  = (float*)alloc(BT * Sc * 4);
  __bf16* binding_bf= (__bf16*)alloc(BT * Sc * 2);
  float*  usage_f   = (float*)alloc(BS * 4);
  float*  write_f   = (float*)alloc(BS * Dc * 4);
  __bf16* wo_bf     = (__bf16*)alloc(BS * Dc * 2);
  float*  write2_f  = (float*)alloc(BS * Dc * 4);
  __bf16* write2_bf = (__bf16*)alloc(BS * Dc * 2);
  __bf16* cat_mw_bf = (__bf16*)alloc(BS * 2 * Dc * 2);
  float*  gate_f    = (float*)alloc(BS * Dc * 4);
  __bf16* rk_bf     = (__bf16*)alloc(BS * Dc * 2);
  __bf16* rv_bf     = (__bf16*)alloc(BS * Dc * 2);
  __bf16* rw_bf     = (__bf16*)alloc(BT * Sc * 2);
  __bf16* ctx_bf    = (__bf16*)alloc(BT * Dc * 2);
  // concat(tokens, context) aliases q_bf+v_bf (dead by then) to bound ws usage
  __bf16* cat_tc_bf = q_bf;                              // BT * 2D bf16
  float*  fused_f   = (float*)alloc(BT * Dc * 4);
  float*  pooled_f  = (float*)alloc((size_t)Bc * Dc * 4);
  (void)ws_size; (void)in_sizes; (void)n_in; (void)out_size;

  const float scale = 1.0f / sqrtf((float)Dc);
  auto EL = [](long n) { return dim3((unsigned)((n + 255) / 256)); };
  auto GG = [](int M, int N, int nb) {   // 32x32 per wave, 8 waves per block
    return dim3((unsigned)((N / 32 + 7) / 8), (unsigned)(M / 32), (unsigned)nb);
  };
  const dim3 thr256(256);

  // 1) weights -> bf16, transposed [N,K]
  struct WT { const float* w; __bf16* t; int K, N; } wts[9] = {
    {q_W, qWt, Dc, Dc}, {k_W, kWt, Dc, Dc}, {v_W, vWt, Dc, Dc},
    {wr_W, wrWt, Dc, Dc}, {rq_W, rqWt, Dc, Dc}, {rk_W, rkWt, Dc, Dc},
    {rv_W, rvWt, Dc, Dc}, {gate_W, gateWt, 2 * Dc, Dc}, {fuse_W, fuseWt, 2 * Dc, Dc}};
  for (int i = 0; i < 9; ++i)
    w_transpose_bf16<<<EL((long)wts[i].K * wts[i].N), thr256, 0, stream>>>(
        wts[i].w, wts[i].t, wts[i].K, wts[i].N);

  // 2) tokens = LN(emb gather)
  embed_ln_kernel<<<dim3((unsigned)BT), thr256, 0, stream>>>(
      input_ids, token_emb, pos_emb, tok_g, tok_bb, tokens_bf);

  // 3) q, v, rq projections
  gemm_wmma_bf16<0><<<GG((int)BT, Dc, 1), thr256, 0, stream>>>(
      tokens_bf, 0, qWt, 0, q_b, nullptr, q_bf, 0, (int)BT, Dc, Dc, 1.0f, 0);
  gemm_wmma_bf16<0><<<GG((int)BT, Dc, 1), thr256, 0, stream>>>(
      tokens_bf, 0, vWt, 0, v_b, nullptr, v_bf, 0, (int)BT, Dc, Dc, 1.0f, 0);
  gemm_wmma_bf16<0><<<GG((int)BT, Dc, 1), thr256, 0, stream>>>(
      tokens_bf, 0, rqWt, 0, rq_b, nullptr, rq_bf, 0, (int)BT, Dc, Dc, 1.0f, 0);

  // 4) memory init
  mem_init_kernel<<<EL(BS * Dc), thr256, 0, stream>>>(memory_init, memory_f,
                                                      memory_bf, BS * Dc);

  // 5) slot refinement steps
  for (int step = 0; step < STEPS; ++step) {
    gemm_wmma_bf16<0><<<GG((int)BS, Dc, 1), thr256, 0, stream>>>(
        memory_bf, 0, kWt, 0, k_b, nullptr, k_bf, 0, (int)BS, Dc, Dc, 1.0f, 0);
    gemm_wmma_bf16<0><<<GG(Tc, Sc, Bc), thr256, 0, stream>>>(
        q_bf, (long)Tc * Dc, k_bf, (long)Sc * Dc, nullptr, scores_f, nullptr,
        (long)Tc * Sc, Tc, Sc, Dc, scale, 1);
    topk_softmax_kernel<<<dim3((unsigned)(BT / 8)), thr256, 0, stream>>>(
        scores_f, amask, binding_bf, BT);
    slot_usage_kernel<<<dim3((unsigned)BS), thr256, 0, stream>>>(binding_bf, usage_f);
    gemm_wmma_bf16<1><<<GG(Sc, Dc, Bc), thr256, 0, stream>>>(
        binding_bf, (long)Tc * Sc, v_bf, (long)Tc * Dc, nullptr, write_f,
        nullptr, (long)Sc * Dc, Sc, Dc, Tc, 1.0f, 0);
    write_over_usage_kernel<<<EL(BS * Dc), thr256, 0, stream>>>(write_f, usage_f,
                                                               wo_bf, BS * Dc);
    gemm_wmma_bf16<0><<<GG((int)BS, Dc, 1), thr256, 0, stream>>>(
        wo_bf, 0, wrWt, 0, wr_b, write2_f, write2_bf, 0, (int)BS, Dc, Dc, 1.0f, 0);
    pack2_bf_kernel<<<EL(BS * 2 * Dc), thr256, 0, stream>>>(memory_bf, write2_bf,
                                                            cat_mw_bf, BS * 2 * Dc);
    gemm_wmma_bf16<0><<<GG((int)BS, Dc, 1), thr256, 0, stream>>>(
        cat_mw_bf, 0, gateWt, 0, gate_b, gate_f, nullptr, 0, (int)BS, Dc,
        2 * Dc, 1.0f, 2);
    mem_update_ln_kernel<<<dim3((unsigned)BS), thr256, 0, stream>>>(
        memory_f, memory_bf, gate_f, write2_f, slot_g, slot_bb);
  }

  // 6) readout
  gemm_wmma_bf16<0><<<GG((int)BS, Dc, 1), thr256, 0, stream>>>(
      memory_bf, 0, rkWt, 0, rk_b, nullptr, rk_bf, 0, (int)BS, Dc, Dc, 1.0f, 0);
  gemm_wmma_bf16<0><<<GG((int)BS, Dc, 1), thr256, 0, stream>>>(
      memory_bf, 0, rvWt, 0, rv_b, nullptr, rv_bf, 0, (int)BS, Dc, Dc, 1.0f, 0);
  gemm_wmma_bf16<0><<<GG(Tc, Sc, Bc), thr256, 0, stream>>>(
      rq_bf, (long)Tc * Dc, rk_bf, (long)Sc * Dc, nullptr, scores_f, nullptr,
      (long)Tc * Sc, Tc, Sc, Dc, scale, 1);
  topk_softmax_kernel<<<dim3((unsigned)(BT / 8)), thr256, 0, stream>>>(
      scores_f, amask, rw_bf, BT);
  gemm_wmma_bf16<2><<<GG(Tc, Dc, Bc), thr256, 0, stream>>>(
      rw_bf, (long)Tc * Sc, rv_bf, (long)Sc * Dc, nullptr, nullptr, ctx_bf,
      (long)Tc * Dc, Tc, Dc, Sc, 1.0f, 0);
  pack2_bf_kernel<<<EL(BT * 2 * Dc), thr256, 0, stream>>>(tokens_bf, ctx_bf,
                                                          cat_tc_bf, BT * 2 * Dc);
  gemm_wmma_bf16<0><<<GG((int)BT, Dc, 1), thr256, 0, stream>>>(
      cat_tc_bf, 0, fuseWt, 0, fuse_b, fused_f, nullptr, 0, (int)BT, Dc,
      2 * Dc, 1.0f, 0);
  pool_kernel<<<dim3((unsigned)(Bc * (Dc / 256))), thr256, 0, stream>>>(
      fused_f, amask, pooled_f);
  logits_kernel<<<dim3(1), dim3(Bc * Cc), 0, stream>>>(pooled_f, cls_W, cls_b,
                                                       logits);
}
